// AttentionAggregator_5806795784352
// MI455X (gfx1250) — matrix operand order repaired and checked by reading, not compiled, not run on hardware
//
#include <hip/hip_runtime.h>

#define IN_DIM  4096
#define OUT_DIM 1024
#define NN      4096
#define MMN     8192
#define NSAMP   10

typedef __attribute__((ext_vector_type(16))) __bf16 v16bf;
typedef __attribute__((ext_vector_type(8)))  float  v8f;
typedef __attribute__((ext_vector_type(4)))  unsigned int u32x4;

// ---------- f32 -> bf16 elementwise convert ----------
__global__ __launch_bounds__(256)
void cvt_f32_bf16(const float* __restrict__ in, __bf16* __restrict__ out, int n) {
    int i = blockIdx.x * 256 + threadIdx.x;
    if (i < n) out[i] = (__bf16)in[i];
}

// ---------- w1[d] = sum_o W[o, d] * a1[o]  (folds node GEMM into a GEMV) ----------
__global__ __launch_bounds__(256)
void compute_w1(const float* __restrict__ W, const float* __restrict__ a1,
                float* __restrict__ w1) {
    int d = blockIdx.x * 256 + threadIdx.x;          // 0 .. IN_DIM-1
    float s = 0.f;
    for (int o = 0; o < OUT_DIM; ++o)
        s += W[(size_t)o * IN_DIM + d] * a1[o];      // coalesced across threads
    w1[d] = s;
}

// ---------- out[row] = dot(X[row, :len], v[:len]) ----------
__global__ __launch_bounds__(256)
void rowdot(const float* __restrict__ X, const float* __restrict__ v,
            float* __restrict__ out, int len) {
    int row = blockIdx.x;
    const float* x = X + (size_t)row * len;
    float s = 0.f;
    for (int d = threadIdx.x; d < len; d += 256) s += x[d] * v[d];
    __shared__ float red[256];
    red[threadIdx.x] = s; __syncthreads();
    for (int off = 128; off > 0; off >>= 1) {
        if (threadIdx.x < off) red[threadIdx.x] += red[threadIdx.x + off];
        __syncthreads();
    }
    if (threadIdx.x == 0) out[row] = red[0];
}

// ---------- neigh_t[MMN, OUT_DIM] = A[MMN, K](bf16) @ B[OUT_DIM, K](bf16)^T ----------
// Per-wave tile: 32(M) x 64(N) -> 8 WMMA per K-step of 32, 12 b128 loads.
// Double-buffered fragment sets overlap VMEM with the matrix pipe.
// 8 waves/block -> 256x64 block tile. grid = (MMN/256, OUT_DIM/64), block = 256.
__global__ __launch_bounds__(256)
void gat_gemm_bf16(const __bf16* __restrict__ A, const __bf16* __restrict__ B,
                   float* __restrict__ C) {
    const int K = IN_DIM;
    const int lane = threadIdx.x & 31;
    const int wave = threadIdx.x >> 5;
    const int l15  = lane & 15;
    const int hi   = lane >> 4;                      // 0 or 1

    const int mrow   = (blockIdx.x * 8 + wave) * 32; // first of 32 rows
    const int ntile0 = blockIdx.y * 4;               // first of 4 16-col tiles

    // A frags: lane holds row; K-chunks [hi*8, +8) and [hi*8+16, +8)
    const __bf16* pA0 = A + (size_t)(mrow + l15) * K + hi * 8;
    const __bf16* pA1 = pA0 + (size_t)16 * K;
    // B frags: lane holds W-row; K-chunks [hi*8, +8) and [hi*8+16, +8)
    const __bf16* pB0 = B + (size_t)((ntile0 + 0) * 16 + l15) * K + hi * 8;
    const __bf16* pB1 = pB0 + (size_t)16 * K;
    const __bf16* pB2 = pB0 + (size_t)32 * K;
    const __bf16* pB3 = pB0 + (size_t)48 * K;

    v8f acc00 = {}, acc01 = {}, acc02 = {}, acc03 = {};
    v8f acc10 = {}, acc11 = {}, acc12 = {}, acc13 = {};

    v16bf A0x, A1x, B0x, B1x, B2x, B3x;   // set 0
    v16bf A0y, A1y, B0y, B1y, B2y, B3y;   // set 1

#define LOADSET(s, ko)                                                    \
    ((u32x4*)&A0##s)[0] = *(const u32x4*)(pA0 + (ko));                    \
    ((u32x4*)&A0##s)[1] = *(const u32x4*)(pA0 + (ko) + 16);               \
    ((u32x4*)&A1##s)[0] = *(const u32x4*)(pA1 + (ko));                    \
    ((u32x4*)&A1##s)[1] = *(const u32x4*)(pA1 + (ko) + 16);               \
    ((u32x4*)&B0##s)[0] = *(const u32x4*)(pB0 + (ko));                    \
    ((u32x4*)&B0##s)[1] = *(const u32x4*)(pB0 + (ko) + 16);               \
    ((u32x4*)&B1##s)[0] = *(const u32x4*)(pB1 + (ko));                    \
    ((u32x4*)&B1##s)[1] = *(const u32x4*)(pB1 + (ko) + 16);               \
    ((u32x4*)&B2##s)[0] = *(const u32x4*)(pB2 + (ko));                    \
    ((u32x4*)&B2##s)[1] = *(const u32x4*)(pB2 + (ko) + 16);               \
    ((u32x4*)&B3##s)[0] = *(const u32x4*)(pB3 + (ko));                    \
    ((u32x4*)&B3##s)[1] = *(const u32x4*)(pB3 + (ko) + 16);

#define WMMASET(s)                                                                                  \
    acc00 = __builtin_amdgcn_wmma_f32_16x16x32_bf16(false, A0##s, false, B0##s, (short)0, acc00, false, false); \
    acc01 = __builtin_amdgcn_wmma_f32_16x16x32_bf16(false, A0##s, false, B1##s, (short)0, acc01, false, false); \
    acc02 = __builtin_amdgcn_wmma_f32_16x16x32_bf16(false, A0##s, false, B2##s, (short)0, acc02, false, false); \
    acc03 = __builtin_amdgcn_wmma_f32_16x16x32_bf16(false, A0##s, false, B3##s, (short)0, acc03, false, false); \
    acc10 = __builtin_amdgcn_wmma_f32_16x16x32_bf16(false, A1##s, false, B0##s, (short)0, acc10, false, false); \
    acc11 = __builtin_amdgcn_wmma_f32_16x16x32_bf16(false, A1##s, false, B1##s, (short)0, acc11, false, false); \
    acc12 = __builtin_amdgcn_wmma_f32_16x16x32_bf16(false, A1##s, false, B2##s, (short)0, acc12, false, false); \
    acc13 = __builtin_amdgcn_wmma_f32_16x16x32_bf16(false, A1##s, false, B3##s, (short)0, acc13, false, false);

    LOADSET(x, 0)                        // set x <- k=0
#pragma unroll 1
    for (int k0 = 0; k0 < K - 64; k0 += 64) {
        LOADSET(y, k0 + 32)              // prefetch while computing set x
        WMMASET(x)
        LOADSET(x, k0 + 64)              // prefetch while computing set y
        WMMASET(y)
    }
    LOADSET(y, K - 32)
    WMMASET(x)
    WMMASET(y)
#undef LOADSET
#undef WMMASET

    // D layout: VGPR r, lane L -> m = r + (L>=16)*8, n = L&15
    float* pC0 = C + (size_t)(mrow + hi * 8) * OUT_DIM + ntile0 * 16 + l15;
    float* pC1 = pC0 + (size_t)16 * OUT_DIM;
#pragma unroll
    for (int r = 0; r < 8; ++r) {
        pC0[(size_t)r * OUT_DIM +  0] = acc00[r];
        pC0[(size_t)r * OUT_DIM + 16] = acc01[r];
        pC0[(size_t)r * OUT_DIM + 32] = acc02[r];
        pC0[(size_t)r * OUT_DIM + 48] = acc03[r];
        pC1[(size_t)r * OUT_DIM +  0] = acc10[r];
        pC1[(size_t)r * OUT_DIM + 16] = acc11[r];
        pC1[(size_t)r * OUT_DIM + 32] = acc12[r];
        pC1[(size_t)r * OUT_DIM + 48] = acc13[r];
    }
}

// ---------- per-node: dedupe 10 indices, leaky-relu+softmax, weighted gather ----------
__global__ __launch_bounds__(256)
void aggregate(const float* __restrict__ neigh_t, const float* __restrict__ s1,
               const float* __restrict__ s2, const int* __restrict__ nidx,
               float* __restrict__ out) {
    const int i = blockIdx.x;

    // Every thread redundantly computes the (tiny, uniform) scalar part.
    int raw[NSAMP];
#pragma unroll
    for (int t = 0; t < NSAMP; ++t) raw[t] = nidx[i * NSAMP + t];

    const float si = s1[i];
    int   ids[NSAMP];
    float w[NSAMP];
    int   cnt = 0;
#pragma unroll
    for (int t = 0; t < NSAMP; ++t) {
        bool dup = false;
#pragma unroll
        for (int u = 0; u < NSAMP; ++u)
            if (u < t && raw[u] == raw[t]) dup = true;
        if (!dup) {
            float e = si + s2[raw[t]];
            e = (e >= 0.f) ? e : 0.2f * e;           // leaky_relu(0.2)
            ids[cnt] = raw[t];
            w[cnt]   = e;
            ++cnt;
        }
    }
    float mx = -3.0e38f;
    for (int c = 0; c < cnt; ++c) mx = fmaxf(mx, w[c]);
    float sum = 0.f;
    for (int c = 0; c < cnt; ++c) { w[c] = expf(w[c] - mx); sum += w[c]; }
    const float inv = 1.f / sum;
    for (int c = 0; c < cnt; ++c) w[c] *= inv;

    // 1024 outputs / 256 threads = exactly one float4 per thread (b128 gathers)
    const int o = threadIdx.x * 4;
    float4 acc = make_float4(0.f, 0.f, 0.f, 0.f);
    for (int c = 0; c < cnt; ++c) {
        const float4 v = *(const float4*)(neigh_t + (size_t)ids[c] * OUT_DIM + o);
        acc.x += w[c] * v.x;
        acc.y += w[c] * v.y;
        acc.z += w[c] * v.z;
        acc.w += w[c] * v.w;
    }
    *(float4*)(out + (size_t)i * OUT_DIM + o) = acc;
}

extern "C" void kernel_launch(void* const* d_in, const int* in_sizes, int n_in,
                              void* d_out, int out_size, void* d_ws, size_t ws_size,
                              hipStream_t stream) {
    const float* node   = (const float*)d_in[0];   // [NN, IN_DIM]
    const float* neigh  = (const float*)d_in[1];   // [MMN, IN_DIM]
    const float* weight = (const float*)d_in[2];   // [OUT_DIM, IN_DIM]
    const float* attv   = (const float*)d_in[3];   // [2*OUT_DIM]
    const int*   nidx   = (const int*)d_in[4];     // [NN, NSAMP]
    float* out = (float*)d_out;

    char* ws = (char*)d_ws;
    __bf16* nb      = (__bf16*)ws; ws += (size_t)MMN * IN_DIM * 2;       // 64 MB
    __bf16* wb      = (__bf16*)ws; ws += (size_t)OUT_DIM * IN_DIM * 2;   //  8 MB
    float*  neigh_t = (float*)ws;  ws += (size_t)MMN * OUT_DIM * 4;      // 32 MB
    float*  w1      = (float*)ws;  ws += (size_t)IN_DIM * 4;
    float*  s1      = (float*)ws;  ws += (size_t)NN * 4;
    float*  s2      = (float*)ws;  ws += (size_t)MMN * 4;

    cvt_f32_bf16<<<(MMN * IN_DIM) / 256, 256, 0, stream>>>(neigh, nb, MMN * IN_DIM);
    cvt_f32_bf16<<<(OUT_DIM * IN_DIM) / 256, 256, 0, stream>>>(weight, wb, OUT_DIM * IN_DIM);

    compute_w1<<<IN_DIM / 256, 256, 0, stream>>>(weight, attv, w1);
    rowdot<<<NN, 256, 0, stream>>>(node, w1, s1, IN_DIM);

    gat_gemm_bf16<<<dim3(MMN / 256, OUT_DIM / 64), 256, 0, stream>>>(nb, wb, neigh_t);

    rowdot<<<MMN, 256, 0, stream>>>(neigh_t, attv + OUT_DIM, s2, OUT_DIM);

    aggregate<<<NN, 256, 0, stream>>>(neigh_t, s1, s2, nidx, out);
}
